// GraphSender_43447889166782
// MI455X (gfx1250) — matrix operand order repaired and checked by reading, not compiled, not run on hardware
//
#include <hip/hip_runtime.h>
#include <hip/hip_bf16.h>
#include <math.h>

// ---- problem constants (match reference) ----
#define FEAT 16
#define EMB  32
#define HID  32
#define VCLS 12
#define EAtt 9
#define NGRP 64

typedef __attribute__((ext_vector_type(16))) _Float16 v16h;
typedef __attribute__((ext_vector_type(8)))  float    v8f;

// ------------------------------------------------------------------
// Pack w2 [32, din*32] into per-lane WMMA B-fragment order:
// out[((i*2+t)*32 + ln)*16 + j] = w2[k, i*32 + t*16 + (ln&15)],
//   k = (j<8) ? (ln>>4)*8 + j : 16 + (ln>>4)*8 + (j-8)   (lane n holds column n)
// ------------------------------------------------------------------
__global__ void pack_w2_kernel(const float* __restrict__ w2,
                               _Float16* __restrict__ out, int din) {
  int tid = blockIdx.x * blockDim.x + threadIdx.x;
  int total = din * 2 * 32 * 16;
  if (tid >= total) return;
  int j  = tid & 15;
  int ln = (tid >> 4) & 31;
  int t  = (tid >> 9) & 1;
  int i  = tid >> 10;
  int k   = (j < 8) ? ((ln >> 4) * 8 + j) : (16 + (ln >> 4) * 8 + (j - 8));
  int col = i * 32 + t * 16 + (ln & 15);
  out[tid] = (_Float16)w2[k * (din * 32) + col];
}

// degree of each destination node (for segment mean)
__global__ void degree_kernel(const int* __restrict__ dst, float* __restrict__ deg, int E) {
  int tid = blockIdx.x * blockDim.x + threadIdx.x;
  if (tid < E) atomicAdd(&deg[dst[tid]], 1.0f);
}

// per-group node counts (for global mean pool)
__global__ void group_count_kernel(const int* __restrict__ batch, float* __restrict__ cnt, int N) {
  int tid = blockIdx.x * blockDim.x + threadIdx.x;
  if (tid < N) atomicAdd(&cnt[batch[tid]], 1.0f);
}

// ------------------------------------------------------------------
// Fused NNConv edge kernel:
//   a    = relu(ea @ w1 + b1)                  (per-lane VALU, 16 vals/lane)
//   msg  = Σ_i (h_src[:,i] ⊙ a) @ Bpk[i]       (WMMA f32<-f16, 2 N-tiles)
//        + h_src @ b2r                         (per-lane VALU via LDS)
//   agg[dst] += msg                            (global_atomic_add_f32)
// One wave = 16 edges. blockDim = 256 (8 waves).
// ------------------------------------------------------------------
template <int DIN>
__launch_bounds__(256)
__global__ void nnconv_edge_kernel(const float* __restrict__ h,     // [N, DIN]
                                   const float* __restrict__ ea,    // [E, 9]
                                   const int*   __restrict__ src,
                                   const int*   __restrict__ dst,
                                   const float* __restrict__ w1,    // [9,32]
                                   const float* __restrict__ b1,    // [32]
                                   const _Float16* __restrict__ Bpk,
                                   const float* __restrict__ b2,    // [DIN*32]
                                   float* __restrict__ agg,         // [N,32]
                                   int E) {
  __shared__ float bias_lds[8][16][32];
  __shared__ int   dst_lds[8][16];

  const int ln = threadIdx.x & 31;
  const int w  = threadIdx.x >> 5;
  const int ebase = (blockIdx.x * 8 + w) * 16;
  const int e_loc = ln & 15;
  int eg = ebase + e_loc;
  int ec = (eg < E) ? eg : (E - 1);   // clamp: keep EXEC all-ones for WMMA

  // gather source-node feature row (lanes l and l+16 share edge e_loc)
  const int s = src[ec];
  float hrow[DIN];
#pragma unroll
  for (int i = 0; i < DIN; ++i) hrow[i] = h[s * DIN + i];

  if (ln < 16) dst_lds[w][ln] = dst[ec];

  // edge attributes
  float eav[EAtt];
#pragma unroll
  for (int j = 0; j < EAtt; ++j) eav[j] = ea[ec * EAtt + j];

  // hidden a[e, k] for exactly the 16 K-slots this lane owns in the
  // WMMA A-fragment layout (16-bit A, 16x32):
  //   lane<16 : K in {kb..kb+7, 16+kb..16+kb+7}, kb = 0
  //   lane>=16: same with kb = 8
  const int kb = (ln >> 4) * 8;
  v16h a_lane;
#pragma unroll
  for (int jj = 0; jj < 16; ++jj) {
    int k = (jj < 8) ? (kb + jj) : (16 + kb + (jj - 8));
    float acc = b1[k];
#pragma unroll
    for (int j = 0; j < EAtt; ++j) acc += eav[j] * w1[j * 32 + k];
    a_lane[jj] = (_Float16)fmaxf(acc, 0.0f);
  }

  // bias term h_src @ b2r into LDS (this lane: its own edge, 16 columns)
  {
    const int ob = (ln >> 4) * 16;
#pragma unroll
    for (int jo = 0; jo < 16; ++jo) {
      int o = ob + jo;
      float acc = 0.0f;
#pragma unroll
      for (int i = 0; i < DIN; ++i) acc += hrow[i] * b2[i * 32 + o];
      bias_lds[w][e_loc][o] = acc;
    }
  }
  __syncthreads();

  // ---- WMMA main loop: DIN chunks of K=32, two 16-col N tiles ----
  const v16h* __restrict__ Bp = (const v16h*)Bpk;
  v8f c0 = {};
  v8f c1 = {};
#pragma unroll
  for (int i = 0; i < DIN; ++i) {
    _Float16 sc = (_Float16)hrow[i];
    v16h as;
#pragma unroll
    for (int jj = 0; jj < 16; ++jj) as[jj] = a_lane[jj] * sc;  // v_pk_mul_f16
    v16h bt0 = Bp[(i * 2 + 0) * 32 + ln];
    v16h bt1 = Bp[(i * 2 + 1) * 32 + ln];
    c0 = __builtin_amdgcn_wmma_f32_16x16x32_f16(false, as, false, bt0,
                                                (short)0, c0, false, false);
    c1 = __builtin_amdgcn_wmma_f32_16x16x32_f16(false, as, false, bt1,
                                                (short)0, c1, false, false);
  }

  // ---- scatter-add messages (C layout: lane=col, vgpr v = row v+8*(ln>>4))
  const int col = ln & 15;
  const int rb  = (ln >> 4) * 8;
#pragma unroll
  for (int v = 0; v < 8; ++v) {
    int er = rb + v;
    if (ebase + er < E) {
      int d = dst_lds[w][er];
      atomicAdd(&agg[d * 32 + col],      c0[v] + bias_lds[w][er][col]);
      atomicAdd(&agg[d * 32 + 16 + col], c1[v] + bias_lds[w][er][col + 16]);
    }
  }
}

// ------------------------------------------------------------------
// Node update: hout = relu(agg/deg + hin @ root + bias)
// ------------------------------------------------------------------
__global__ void node_update_kernel(const float* __restrict__ agg,
                                   const float* __restrict__ deg,
                                   const float* __restrict__ hin,
                                   const float* __restrict__ root,
                                   const float* __restrict__ bias,
                                   float* __restrict__ hout,
                                   int N, int din) {
  int tid = blockIdx.x * blockDim.x + threadIdx.x;
  if (tid >= N * 32) return;
  int n = tid >> 5, o = tid & 31;
  float acc = agg[tid] / fmaxf(deg[n], 1.0f) + bias[o];
  for (int i = 0; i < din; ++i) acc += hin[n * din + i] * root[i * 32 + o];
  hout[tid] = fmaxf(acc, 0.0f);
}

// same, but fuses the global mean-pool accumulation (no h2 buffer needed)
__global__ void node_update_pool_kernel(const float* __restrict__ agg,
                                        const float* __restrict__ deg,
                                        const float* __restrict__ hin,
                                        const float* __restrict__ root,
                                        const float* __restrict__ bias,
                                        const int* __restrict__ batch,
                                        float* __restrict__ pool,
                                        int N, int din) {
  int tid = blockIdx.x * blockDim.x + threadIdx.x;
  if (tid >= N * 32) return;
  int n = tid >> 5, o = tid & 31;
  float acc = agg[tid] / fmaxf(deg[n], 1.0f) + bias[o];
  for (int i = 0; i < din; ++i) acc += hin[n * din + i] * root[i * 32 + o];
  atomicAdd(&pool[batch[n] * 32 + o], fmaxf(acc, 0.0f));
}

// ------------------------------------------------------------------
// Head: pooled mean -> linear -> log_softmax. One thread per group.
// ------------------------------------------------------------------
__global__ void head_kernel(const float* __restrict__ pool,
                            const float* __restrict__ cnt,
                            const float* __restrict__ lw,   // [32,12]
                            const float* __restrict__ lb,   // [12]
                            float* __restrict__ out) {
  int g = threadIdx.x;
  if (g >= NGRP) return;
  float inv = 1.0f / fmaxf(cnt[g], 1.0f);
  float p[HID];
#pragma unroll
  for (int o = 0; o < HID; ++o) p[o] = pool[g * HID + o] * inv;
  float logits[VCLS];
  float mx = -1e30f;
#pragma unroll
  for (int v = 0; v < VCLS; ++v) {
    float a = lb[v];
#pragma unroll
    for (int o = 0; o < HID; ++o) a += p[o] * lw[o * VCLS + v];
    logits[v] = a;
    mx = fmaxf(mx, a);
  }
  float sum = 0.0f;
#pragma unroll
  for (int v = 0; v < VCLS; ++v) sum += expf(logits[v] - mx);
  float lse = mx + logf(sum);
#pragma unroll
  for (int v = 0; v < VCLS; ++v) out[g * VCLS + v] = logits[v] - lse;
}

// ------------------------------------------------------------------
extern "C" void kernel_launch(void* const* d_in, const int* in_sizes, int n_in,
                              void* d_out, int out_size, void* d_ws, size_t ws_size,
                              hipStream_t stream) {
  const float* x    = (const float*)d_in[0];
  const float* ea   = (const float*)d_in[1];
  const int*   eidx = (const int*)d_in[2];
  const int*   batch= (const int*)d_in[3];
  const float* n1w1 = (const float*)d_in[4];
  const float* n1b1 = (const float*)d_in[5];
  const float* n1w2 = (const float*)d_in[6];
  const float* n1b2 = (const float*)d_in[7];
  const float* root1= (const float*)d_in[8];
  const float* bias1= (const float*)d_in[9];
  const float* n2w1 = (const float*)d_in[10];
  const float* n2b1 = (const float*)d_in[11];
  const float* n2w2 = (const float*)d_in[12];
  const float* n2b2 = (const float*)d_in[13];
  const float* root2= (const float*)d_in[14];
  const float* bias2= (const float*)d_in[15];
  const float* lw   = (const float*)d_in[16];
  const float* lb   = (const float*)d_in[17];

  const int N = in_sizes[0] / FEAT;
  const int E = in_sizes[2] / 2;
  const int* src = eidx;
  const int* dst = eidx + E;

  // ---- workspace carve-up ----
  char* ws = (char*)d_ws;
  size_t off = 0;
  auto carve = [&](size_t bytes) { char* p = ws + off; off += (bytes + 255) & ~(size_t)255; return p; };
  _Float16* Bpk1 = (_Float16*)carve((size_t)FEAT * 2 * 32 * 16 * sizeof(_Float16)); // 32 KB
  _Float16* Bpk2 = (_Float16*)carve((size_t)EMB  * 2 * 32 * 16 * sizeof(_Float16)); // 64 KB
  float* deg  = (float*)carve((size_t)N * sizeof(float));
  float* agg  = (float*)carve((size_t)N * 32 * sizeof(float));
  float* h1   = (float*)carve((size_t)N * 32 * sizeof(float));
  float* pool = (float*)carve((size_t)NGRP * HID * sizeof(float));
  float* cnt  = (float*)carve((size_t)NGRP * sizeof(float));
  (void)ws_size; (void)n_in; (void)out_size;

  // ---- zero the accumulators ----
  hipMemsetAsync(deg,  0, (size_t)N * sizeof(float), stream);
  hipMemsetAsync(agg,  0, (size_t)N * 32 * sizeof(float), stream);
  hipMemsetAsync(pool, 0, (size_t)NGRP * HID * sizeof(float), stream);
  hipMemsetAsync(cnt,  0, (size_t)NGRP * sizeof(float), stream);

  // ---- pack WMMA B fragments ----
  pack_w2_kernel<<<(FEAT * 1024 + 255) / 256, 256, 0, stream>>>(n1w2, Bpk1, FEAT);
  pack_w2_kernel<<<(EMB  * 1024 + 255) / 256, 256, 0, stream>>>(n2w2, Bpk2, EMB);

  // ---- degrees and group counts ----
  degree_kernel<<<(E + 255) / 256, 256, 0, stream>>>(dst, deg, E);
  group_count_kernel<<<(N + 255) / 256, 256, 0, stream>>>(batch, cnt, N);

  const int convBlocks = (E + 127) / 128;  // 8 waves x 16 edges per block

  // ---- conv1 ----
  nnconv_edge_kernel<FEAT><<<convBlocks, 256, 0, stream>>>(
      x, ea, src, dst, n1w1, n1b1, Bpk1, n1b2, agg, E);
  node_update_kernel<<<(N * 32 + 255) / 256, 256, 0, stream>>>(
      agg, deg, x, root1, bias1, h1, N, FEAT);

  // ---- conv2 (reuse agg buffer) ----
  hipMemsetAsync(agg, 0, (size_t)N * 32 * sizeof(float), stream);
  nnconv_edge_kernel<EMB><<<convBlocks, 256, 0, stream>>>(
      h1, ea, src, dst, n2w1, n2b1, Bpk2, n2b2, agg, E);
  node_update_pool_kernel<<<(N * 32 + 255) / 256, 256, 0, stream>>>(
      agg, deg, h1, root2, bias2, batch, pool, N, EMB);

  // ---- head ----
  head_kernel<<<1, NGRP, 0, stream>>>(pool, cnt, lw, lb, (float*)d_out);
}
